// TtAttention_49709951484103
// MI455X (gfx1250) — matrix operand order repaired and checked by reading, not compiled, not run on hardware
//
#include <hip/hip_runtime.h>

#define DEVINL __device__ __forceinline__

constexpr int SEQ  = 2048;
constexpr int DIM  = 4096;
constexpr int NH   = 32;
constexpr int NKVH = 8;
constexpr int HDIM = 128;
constexpr int KVD  = NKVH * HDIM;   // 1024
constexpr int WIN  = 1024;
constexpr float SM_SCALE = 0.08838834764831845f; // 1/sqrt(128)

typedef __attribute__((ext_vector_type(16))) __bf16 v16bf;
typedef __attribute__((ext_vector_type(8)))  __bf16 v8bf;
typedef __attribute__((ext_vector_type(8)))  float  v8f;
typedef unsigned int u32x4 __attribute__((ext_vector_type(4)));
typedef int          i32x8 __attribute__((ext_vector_type(8)));
typedef int          i32x4 __attribute__((ext_vector_type(4)));

#if defined(__HIP_DEVICE_COMPILE__) && __has_builtin(__builtin_amdgcn_tensor_load_to_lds)
#define HAVE_TDM 1
#else
#define HAVE_TDM 0
#endif

DEVINL v8f wmma_bf16(v16bf a, v16bf b, v8f c) {
  // D(16x16,f32) = A(16x32,bf16) * B(32x16,bf16) + C
  return __builtin_amdgcn_wmma_f32_16x16x32_bf16(false, a, false, b, (short)0, c,
                                                 false, false);
}

// A-matrix 16x32 bf16 fragment (contiguous K, matching B):
// lanes 0-15 (hf=0): K = 0..15 (elems 0..15)
// lanes 16-31 (hf=1): K = 16..31
DEVINL v16bf load_a_frag(const __bf16* row, int hf) {
  v8bf lo = *reinterpret_cast<const v8bf*>(row + hf * 16);
  v8bf hi = *reinterpret_cast<const v8bf*>(row + hf * 16 + 8);
  v16bf a;
#pragma unroll
  for (int i = 0; i < 8; ++i) { a[i] = lo[i]; a[i + 8] = hi[i]; }
  return a;
}

// B-matrix 32x16 bf16 fragment: lane holds col N=lane&15, K = hf*16 .. hf*16+15
// (two 16B loads so only 16B alignment is required; works for LDS and global)
DEVINL v16bf load_b_frag(const __bf16* colk) {
  v8bf lo = *reinterpret_cast<const v8bf*>(colk);
  v8bf hi = *reinterpret_cast<const v8bf*>(colk + 8);
  v16bf b;
#pragma unroll
  for (int i = 0; i < 8; ++i) { b[i] = lo[i]; b[i + 8] = hi[i]; }
  return b;
}

DEVINL unsigned lds_byte_off(const void* p) {
  // generic LDS address: addr[31:0] is the LDS byte offset (ISA 10.2 aperture rule)
  return (unsigned)(unsigned long long)p;
}

#if HAVE_TDM
// TDM 2D tile load: global (row-major, rowStride elems of bf16) -> LDS.
// LDS rows are padded: +4 DWORDs every 32 DWORDs (row stride 144B for 128B rows).
DEVINL void tdm_load_2d(unsigned lds_off, const __bf16* gptr, int tile_rows,
                        int tile_cols, int row_stride_elems) {
  const unsigned long long ga = (unsigned long long)gptr;
  u32x4 g0;
  g0[0] = 1u;                                   // count=1 (user descriptor)
  g0[1] = lds_off;                              // lds_addr (bytes)
  g0[2] = (unsigned)(ga & 0xFFFFFFFFu);         // global_addr[31:0]
  g0[3] = (unsigned)((ga >> 32) & 0x1FFFFFFu)   // global_addr[56:32]
        | (2u << 30);                           // type=2 ("image")
  i32x8 g1;
  // data_size=1 (2B) | pad_enable | pad_interval=4 (32 DW) | pad_amount=3 (4 DW)
  g1[0] = (int)((1u << 16) | (1u << 20) | (4u << 22) | (3u << 25));
  g1[1] = (int)(((unsigned)tile_cols & 0xFFFFu) << 16);          // tensor_dim0[15:0]
  g1[2] = (int)((((unsigned)tile_cols) >> 16) |
                (((unsigned)tile_rows & 0xFFFFu) << 16));        // dim0 hi | dim1 lo
  g1[3] = (int)((((unsigned)tile_rows) >> 16) |
                (((unsigned)tile_cols & 0xFFFFu) << 16));        // dim1 hi | tile_dim0
  g1[4] = (int)((unsigned)tile_rows & 0xFFFFu);                  // tile_dim1 | tile_dim2=0
  g1[5] = row_stride_elems;                                      // tensor_dim0_stride[31:0]
  g1[6] = 0;                                                     // stride hi | dim1_stride lo
  g1[7] = 0;
  const i32x4 z4 = {0, 0, 0, 0};
  const i32x8 z8 = {0, 0, 0, 0, 0, 0, 0, 0};
  // 6-arg form (clang-23 / therock-10.0 headers)
  __builtin_amdgcn_tensor_load_to_lds(g0, g1, z4, z4, z8, 0);
}
#endif

// ---------------- fp32 -> bf16 conversion ----------------
__global__ void cvt_bf16(const float* __restrict__ s, __bf16* __restrict__ d, size_t n) {
  for (size_t i = blockIdx.x * (size_t)blockDim.x + threadIdx.x; i < n;
       i += (size_t)gridDim.x * blockDim.x)
    d[i] = (__bf16)s[i];
}

// ---------------- TN GEMM: C[M,N] = A[M,K] * B[N,K]^T ----------------
// Block 256 thr = 8 waves; block tile 128x64; TDM double-buffered LDS staging.
constexpr int LROW = 72;  // 64 bf16 + 4 DWORD TDM pad = 144B LDS row stride

template <bool F32OUT>
__global__ __launch_bounds__(256) void gemm_tn(const __bf16* __restrict__ A,
                                               const __bf16* __restrict__ B,
                                               void* __restrict__ Cv,
                                               int M, int N, int K) {
  __shared__ __bf16 As[2][128][LROW];
  __shared__ __bf16 Bs[2][64][LROW];
  (void)M;

  const int lane = threadIdx.x & 31;
  const int wid  = threadIdx.x >> 5;
  const int hf   = lane >> 4;
  const int l15  = lane & 15;
  const int wr   = wid >> 1;          // 0..3
  const int wc   = wid & 1;           // 0..1
  const int rowBase = blockIdx.y * 128;
  const int colBase = blockIdx.x * 64;

  const __bf16* Ag = A + (size_t)rowBase * K;
  const __bf16* Bg = B + (size_t)colBase * K;

  v8f acc00 = {}, acc01 = {}, acc10 = {}, acc11 = {};
  const int nIter = K / 64;

#if HAVE_TDM
  unsigned asOff[2] = {lds_byte_off(&As[0][0][0]), lds_byte_off(&As[1][0][0])};
  unsigned bsOff[2] = {lds_byte_off(&Bs[0][0][0]), lds_byte_off(&Bs[1][0][0])};
  if (wid == 0) {
    tdm_load_2d(asOff[0], Ag, 128, 64, K);
    tdm_load_2d(bsOff[0], Bg, 64, 64, K);
  }
#endif

  for (int it = 0; it < nIter; ++it) {
#if HAVE_TDM
    const int b = it & 1;
    if (wid == 0) {
      if (it + 1 < nIter) {
        const int k0n = (it + 1) * 64;
        tdm_load_2d(asOff[b ^ 1], Ag + k0n, 128, 64, K);
        tdm_load_2d(bsOff[b ^ 1], Bg + k0n, 64, 64, K);
        __builtin_amdgcn_s_wait_tensorcnt(2);   // previous buffer's 2 ops done
      } else {
        __builtin_amdgcn_s_wait_tensorcnt(0);
      }
    }
    __syncthreads();
#else
    const int b = 0;
    const int k0 = it * 64;
    __syncthreads();
    // cooperative fallback staging (padded rows)
    for (int idx = threadIdx.x; idx < 128 * 8; idx += 256) {
      const int r = idx >> 3, c = (idx & 7) * 8;
      *reinterpret_cast<v8bf*>(&As[0][r][c]) =
          *reinterpret_cast<const v8bf*>(Ag + (size_t)r * K + k0 + c);
    }
    for (int idx = threadIdx.x; idx < 64 * 8; idx += 256) {
      const int r = idx >> 3, c = (idx & 7) * 8;
      *reinterpret_cast<v8bf*>(&Bs[0][r][c]) =
          *reinterpret_cast<const v8bf*>(Bg + (size_t)r * K + k0 + c);
    }
    __syncthreads();
#endif

#pragma unroll
    for (int kk = 0; kk < 64; kk += 32) {
      v16bf a0 = load_a_frag(&As[b][wr * 32 + l15][kk], hf);
      v16bf a1 = load_a_frag(&As[b][wr * 32 + 16 + l15][kk], hf);
      v16bf b0 = load_b_frag(&Bs[b][wc * 32 + l15][kk + hf * 16]);
      v16bf b1 = load_b_frag(&Bs[b][wc * 32 + 16 + l15][kk + hf * 16]);
      acc00 = wmma_bf16(a0, b0, acc00);
      acc01 = wmma_bf16(a0, b1, acc01);
      acc10 = wmma_bf16(a1, b0, acc10);
      acc11 = wmma_bf16(a1, b1, acc11);
    }
    __syncthreads();   // reads done before next TDM overwrites this buffer
  }

  auto store_tile = [&](const v8f& acc, int mt, int nt) {
#pragma unroll
    for (int r = 0; r < 8; ++r) {
      const int row = rowBase + wr * 32 + mt * 16 + hf * 8 + r;
      const int col = colBase + wc * 32 + nt * 16 + l15;
      if constexpr (F32OUT)
        reinterpret_cast<float*>(Cv)[(size_t)row * N + col] = acc[r];
      else
        reinterpret_cast<__bf16*>(Cv)[(size_t)row * N + col] = (__bf16)acc[r];
    }
  };
  store_tile(acc00, 0, 0);
  store_tile(acc01, 0, 1);
  store_tile(acc10, 1, 0);
  store_tile(acc11, 1, 1);
}

// ---------------- interleaved RoPE, in place on bf16 [S, Hx, 128] ----------------
__global__ void rope_kernel(__bf16* __restrict__ t, const float* __restrict__ cosb,
                            const float* __restrict__ sinb, int Hx, int total) {
  for (int idx = blockIdx.x * blockDim.x + threadIdx.x; idx < total;
       idx += gridDim.x * blockDim.x) {
    const int i  = idx & 63;
    const int sh = idx >> 6;
    const int s  = sh / Hx;
    __bf16* p = t + (size_t)sh * HDIM + 2 * i;
    const float a  = (float)p[0];
    const float b  = (float)p[1];
    const float c  = cosb[s * 64 + i];
    const float sn = sinb[s * 64 + i];
    p[0] = (__bf16)(a * c - b * sn);
    p[1] = (__bf16)(a * sn + b * c);
  }
}

// ---------------- V transpose: [S, KVH, HD] -> [KVH, HD, S] ----------------
__global__ void vtrans_kernel(const __bf16* __restrict__ v, __bf16* __restrict__ vt) {
  const int total = SEQ * KVD;
  for (int idx = blockIdx.x * blockDim.x + threadIdx.x; idx < total;
       idx += gridDim.x * blockDim.x) {
    const int d   = idx % HDIM;
    const int rem = idx / HDIM;
    const int h   = rem % NKVH;
    const int s   = rem / NKVH;
    vt[((size_t)h * HDIM + d) * SEQ + s] = v[idx];
  }
}

// ---------------- sliding-window flash attention ----------------
// Q: [S, NH, HD], K: [S, NKVH, HD] (post-rope), Vt: [NKVH, HD, S], O: [S, NH, HD].
// One wave = 16 query rows; block = 8 waves = 128 queries; grid = (S/128, NH).
__global__ __launch_bounds__(256) void flash_attn(const __bf16* __restrict__ Q,
                                                  const __bf16* __restrict__ Km,
                                                  const __bf16* __restrict__ Vt,
                                                  __bf16* __restrict__ O) {
  __shared__ __bf16 plds[8][16 * 32];   // per-wave P staging tile

  const int lane = threadIdx.x & 31;
  const int wid  = threadIdx.x >> 5;
  const int hf   = lane >> 4;
  const int l15  = lane & 15;
  const int h    = blockIdx.y;
  const int kvh  = h >> 2;              // GQA 4:1
  const int q0   = blockIdx.x * 128 + wid * 16;

  const __bf16* qrow = Q + ((size_t)(q0 + l15) * NH + h) * HDIM;
  v16bf qf[4];
#pragma unroll
  for (int f = 0; f < 4; ++f) qf[f] = load_a_frag(qrow + f * 32, hf);

  const v8f vzero = {};
  v8f o[8];
#pragma unroll
  for (int i = 0; i < 8; ++i) o[i] = vzero;
  float mrun[8], lrun[8];
#pragma unroll
  for (int r = 0; r < 8; ++r) { mrun[r] = -3.0e38f; lrun[r] = 0.0f; }

  int kbeg = q0 - (WIN - 1);
  if (kbeg < 0) kbeg = 0;
  kbeg &= ~31;
  const int kend = q0 + 16;

  for (int kb = kbeg; kb < kend; kb += 32) {
    // prefetch next key/value tiles while computing this one
    if (kb + 64 <= kend) {
      __builtin_prefetch(Km + ((size_t)(kb + 32 + l15) * NKVH + kvh) * HDIM, 0, 1);
      __builtin_prefetch(Vt + ((size_t)kvh * HDIM + l15) * SEQ + kb + 32, 0, 1);
    }

    // ---- scores: S0 = Q*K[kb..kb+15]^T, S1 = Q*K[kb+16..kb+31]^T
    v8f s0 = {}, s1 = {};
#pragma unroll
    for (int f = 0; f < 4; ++f) {
      const __bf16* k0p =
          Km + ((size_t)(kb + l15) * NKVH + kvh) * HDIM + f * 32 + hf * 16;
      const __bf16* k1p =
          Km + ((size_t)(kb + 16 + l15) * NKVH + kvh) * HDIM + f * 32 + hf * 16;
      s0 = wmma_bf16(qf[f], load_b_frag(k0p), s0);
      s1 = wmma_bf16(qf[f], load_b_frag(k1p), s1);
    }

    // ---- mask + online softmax
    float p0[8], p1[8];
#pragma unroll
    for (int r = 0; r < 8; ++r) {
      const int q   = q0 + r + 8 * hf;
      const int kc0 = kb + l15;
      const int kc1 = kb + 16 + l15;
      float t0 = s0[r] * SM_SCALE + ((kc0 <= q && (q - kc0) < WIN) ? 0.0f : -1.0e9f);
      float t1 = s1[r] * SM_SCALE + ((kc1 <= q && (q - kc1) < WIN) ? 0.0f : -1.0e9f);
      float mx = fmaxf(t0, t1);
#pragma unroll
      for (int off = 1; off < 16; off <<= 1) mx = fmaxf(mx, __shfl_xor(mx, off, 32));
      const float mnew = fmaxf(mrun[r], mx);
      const float corr = __expf(mrun[r] - mnew);
      mrun[r] = mnew;
      p0[r] = __expf(t0 - mnew);
      p1[r] = __expf(t1 - mnew);
      float rs = p0[r] + p1[r];
#pragma unroll
      for (int off = 1; off < 16; off <<= 1) rs += __shfl_xor(rs, off, 32);
      lrun[r] = lrun[r] * corr + rs;
#pragma unroll
      for (int fo = 0; fo < 8; ++fo) o[fo][r] *= corr;
    }

    // ---- stage P (C-layout) to LDS, reload as A-layout 16x32 fragment
#pragma unroll
    for (int r = 0; r < 8; ++r) {
      const int row = r + 8 * hf;
      plds[wid][row * 32 + l15]      = (__bf16)p0[r];
      plds[wid][row * 32 + 16 + l15] = (__bf16)p1[r];
    }
    asm volatile("s_wait_dscnt 0x0" ::: "memory");  // same-wave LDS store->load
    v16bf pf = load_a_frag(&plds[wid][l15 * 32], hf);

    // ---- O += P(16x32) * V(32 keys x 128 hd)
#pragma unroll
    for (int fo = 0; fo < 8; ++fo) {
      const __bf16* vp =
          Vt + ((size_t)kvh * HDIM + fo * 16 + l15) * SEQ + kb + hf * 16;
      o[fo] = wmma_bf16(pf, load_b_frag(vp), o[fo]);
    }
  }

  float inv[8];
#pragma unroll
  for (int r = 0; r < 8; ++r) inv[r] = 1.0f / lrun[r];
#pragma unroll
  for (int fo = 0; fo < 8; ++fo) {
#pragma unroll
    for (int r = 0; r < 8; ++r) {
      const int q = q0 + r + 8 * hf;
      O[((size_t)q * NH + h) * HDIM + fo * 16 + l15] = (__bf16)(o[fo][r] * inv[r]);
    }
  }
}

// ---------------- host-side orchestration ----------------
extern "C" void kernel_launch(void* const* d_in, const int* in_sizes, int n_in,
                              void* d_out, int out_size, void* d_ws, size_t ws_size,
                              hipStream_t stream) {
  (void)in_sizes; (void)n_in; (void)out_size; (void)ws_size;
  const float* x  = (const float*)d_in[0];
  const float* wq = (const float*)d_in[1];
  const float* wk = (const float*)d_in[2];
  const float* wv = (const float*)d_in[3];
  const float* wo = (const float*)d_in[4];
  const float* fc = (const float*)d_in[5];
  const float* fs = (const float*)d_in[6];

  char* p = (char*)d_ws;
  auto take = [&](size_t elems) {
    __bf16* r = (__bf16*)p;
    p += elems * sizeof(__bf16);
    return r;
  };
  __bf16* xb  = take((size_t)SEQ * DIM);
  __bf16* wqb = take((size_t)DIM * DIM);
  __bf16* wkb = take((size_t)KVD * DIM);
  __bf16* wvb = take((size_t)KVD * DIM);
  __bf16* wob = take((size_t)DIM * DIM);
  __bf16* qb  = take((size_t)SEQ * DIM);
  __bf16* kb  = take((size_t)SEQ * KVD);
  __bf16* vb  = take((size_t)SEQ * KVD);
  __bf16* vtb = take((size_t)SEQ * KVD);
  __bf16* aob = take((size_t)SEQ * DIM);

  const dim3 blk(256);

  cvt_bf16<<<4096, blk, 0, stream>>>(x,  xb,  (size_t)SEQ * DIM);
  cvt_bf16<<<4096, blk, 0, stream>>>(wq, wqb, (size_t)DIM * DIM);
  cvt_bf16<<<2048, blk, 0, stream>>>(wk, wkb, (size_t)KVD * DIM);
  cvt_bf16<<<2048, blk, 0, stream>>>(wv, wvb, (size_t)KVD * DIM);
  cvt_bf16<<<4096, blk, 0, stream>>>(wo, wob, (size_t)DIM * DIM);

  gemm_tn<false><<<dim3(DIM / 64, SEQ / 128), blk, 0, stream>>>(xb, wqb, qb, SEQ, DIM, DIM);
  gemm_tn<false><<<dim3(KVD / 64, SEQ / 128), blk, 0, stream>>>(xb, wkb, kb, SEQ, KVD, DIM);
  gemm_tn<false><<<dim3(KVD / 64, SEQ / 128), blk, 0, stream>>>(xb, wvb, vb, SEQ, KVD, DIM);

  rope_kernel<<<2048, blk, 0, stream>>>(qb, fc, fs, NH,   SEQ * NH   * (HDIM / 2));
  rope_kernel<<<512,  blk, 0, stream>>>(kb, fc, fs, NKVH, SEQ * NKVH * (HDIM / 2));

  vtrans_kernel<<<2048, blk, 0, stream>>>(vb, vtb);

  flash_attn<<<dim3(SEQ / 128, NH), blk, 0, stream>>>(qb, kb, vtb, aob);

  gemm_tn<true><<<dim3(DIM / 64, SEQ / 128), blk, 0, stream>>>(aob, wob, (float*)d_out,
                                                               SEQ, DIM, DIM);
}
